// IQARegression_69114613730486
// MI455X (gfx1250) — matrix operand order repaired and checked by reading, not compiled, not run on hardware
//
#include <hip/hip_runtime.h>
#include <hip/hip_bf16.h>
#include <math.h>

// ---------------------------------------------------------------------------
// Types for CDNA5 WMMA
// ---------------------------------------------------------------------------
typedef _Float16 half_t;
typedef __attribute__((ext_vector_type(16))) _Float16 v16h;
typedef __attribute__((ext_vector_type(8)))  _Float16 v8h;
typedef __attribute__((ext_vector_type(8)))  float    v8f;

// Problem constants (match reference)
#define BB   8
#define SS   785
#define SP   800          // S padded to mult of 32 for attention K loop
#define DD   128
#define HH   8
#define DH   64
#define DF   512
#define LL   6
#define PIX  784          // 28*28
#define CIN  512
#define MTOK (BB*SS)      // 6280

static __device__ inline v8f wmma_f16(v16h a, v16h b, v8f c) {
  // D = A(16x32,f16) * B(32x16,f16) + C(16x16,f32)
  return __builtin_amdgcn_wmma_f32_16x16x32_f16(false, a, false, b, (short)0, c,
                                                false, false);
}

// A-fragment: 16x32 f16. lane l: row m = l&15; halves h = K[g*16 + h]
// with g = l>>4 (contiguous, matches load_bfrag).
static __device__ inline v16h load_afrag(const half_t* __restrict__ row, int kb, int lane) {
  int lo = (lane >> 4) * 16;
  v8h c0 = *(const v8h*)(row + kb + lo);
  v8h c1 = *(const v8h*)(row + kb + lo + 8);
  return __builtin_shufflevector(c0, c1, 0,1,2,3,4,5,6,7,8,9,10,11,12,13,14,15);
}

// B-fragment: 32x16 f16 from Bt stored row-major [N][K]. lane l: col n = l&15,
// halves h = K[(l>>4)*16 + h]. Contiguous 32B load.
static __device__ inline v16h load_bfrag(const half_t* __restrict__ btrow, int kb, int lane) {
  return *(const v16h*)(btrow + kb + ((lane >> 4) * 16));
}

// ---------------------------------------------------------------------------
// Generic WMMA GEMM: C[M,N] = act(A[M,K] * B[K,N] + bias)
// A: f16 row-major [M,K]; Bt: f16 row-major [N,K] (i.e. B transposed)
// Register-blocked: one wave computes a 16x64 strip (A-fragment reused across
// 4 B-tiles -> 4 accumulators), raising WMMA work per fragment load.
// omode: 0 -> f32 [M,N]; 1 -> f16 [M,N]; 2 -> f16 q/k layout [B,H,S,DH];
//        3 -> f16 v-transposed layout [B,H,DH,SP]
// act:   0 none, 1 exact GELU, 2 ReLU
// block = (32,4): 4 waves, each owns a 64-column strip; grid=(MT, ceil(N/256))
// ---------------------------------------------------------------------------
__global__ void gemm_wmma_kernel(const half_t* __restrict__ A,
                                 const half_t* __restrict__ Bt,
                                 const float*  __restrict__ bias,
                                 float* __restrict__ out32,
                                 half_t* __restrict__ out16,
                                 int M, int N, int K, int act, int omode) {
  const int lane = threadIdx.x;
  const int tm   = blockIdx.x;
  const int cbase = (blockIdx.y * 4 + threadIdx.y) * 64;   // 64-col strip
  if (cbase >= N) return;

  const int ml  = lane & 15;
  const int grp = lane >> 4;

  int arow = tm * 16 + ml;
  if (arow > M - 1) arow = M - 1;          // clamp (stores are guarded)
  const half_t* __restrict__ ap = A + (size_t)arow * K;
  const half_t* __restrict__ bp0 = Bt + (size_t)(cbase +      ml) * K;
  const half_t* __restrict__ bp1 = Bt + (size_t)(cbase + 16 + ml) * K;
  const half_t* __restrict__ bp2 = Bt + (size_t)(cbase + 32 + ml) * K;
  const half_t* __restrict__ bp3 = Bt + (size_t)(cbase + 48 + ml) * K;

  v8f acc0 = {}, acc1 = {}, acc2 = {}, acc3 = {};
  for (int kb = 0; kb < K; kb += 32) {
    v16h af = load_afrag(ap, kb, lane);            // loaded once, used 4x
    acc0 = wmma_f16(af, load_bfrag(bp0, kb, lane), acc0);
    acc1 = wmma_f16(af, load_bfrag(bp1, kb, lane), acc1);
    acc2 = wmma_f16(af, load_bfrag(bp2, kb, lane), acc2);
    acc3 = wmma_f16(af, load_bfrag(bp3, kb, lane), acc3);
  }

  v8f accs[4] = {acc0, acc1, acc2, acc3};
#pragma unroll
  for (int nt = 0; nt < 4; ++nt) {
    const int n = cbase + nt * 16 + ml;            // lane's output column
    const float bz = bias ? bias[n] : 0.f;
#pragma unroll
    for (int r = 0; r < 8; ++r) {
      int m = tm * 16 + grp * 8 + r;
      if (m >= M) continue;
      float v = accs[nt][r] + bz;
      if (act == 1)      v = 0.5f * v * (1.f + erff(v * 0.70710678118f));
      else if (act == 2) v = fmaxf(v, 0.f);
      if (omode == 0) {
        out32[(size_t)m * N + n] = v;
      } else if (omode == 1) {
        out16[(size_t)m * N + n] = (half_t)v;
      } else if (omode == 2) {               // q/k: [B,H,S,DH]
        int b = m / SS, s = m % SS, h = n >> 6, dh = n & 63;
        out16[(((size_t)(b * HH + h) * SS) + s) * DH + dh] = (half_t)v;
      } else {                               // vT: [B,H,DH,SP]
        int b = m / SS, s = m % SS, h = n >> 6, dh = n & 63;
        out16[((size_t)(b * HH + h) * DH + dh) * SP + s] = (half_t)v;
      }
    }
  }
}

// ---------------------------------------------------------------------------
// Conv stem GEMM: out[b*784+p, d] = relu(BN(sum_c xT[b*784+p, c] * Wt[d, c]))
// plus global max via atomicMax on float bits (valid: outputs >= 0)
// Same 16x64-per-wave register blocking (N=128 -> 2 active waves per block).
// ---------------------------------------------------------------------------
__global__ void stem_gemm_kernel(const half_t* __restrict__ A,   // [B*784, 512]
                                 const half_t* __restrict__ Bt,  // conv_w f16 [128,512]
                                 const float* __restrict__ gamma,
                                 const float* __restrict__ beta,
                                 const float* __restrict__ mean,
                                 const float* __restrict__ var,
                                 float* __restrict__ out,        // [B*784, 128]
                                 unsigned int* __restrict__ maxbits) {
  const int lane = threadIdx.x;
  const int tm = blockIdx.x;
  const int cbase = threadIdx.y * 64;
  const int M = BB * PIX, N = DD, K = CIN;
  if (cbase >= N) return;
  const int ml = lane & 15, grp = lane >> 4;

  const half_t* ap  = A  + (size_t)(tm * 16 + ml) * K;
  const half_t* bp0 = Bt + (size_t)(cbase +      ml) * K;
  const half_t* bp1 = Bt + (size_t)(cbase + 16 + ml) * K;
  const half_t* bp2 = Bt + (size_t)(cbase + 32 + ml) * K;
  const half_t* bp3 = Bt + (size_t)(cbase + 48 + ml) * K;

  v8f acc0 = {}, acc1 = {}, acc2 = {}, acc3 = {};
  for (int kb = 0; kb < K; kb += 32) {
    v16h af = load_afrag(ap, kb, lane);
    acc0 = wmma_f16(af, load_bfrag(bp0, kb, lane), acc0);
    acc1 = wmma_f16(af, load_bfrag(bp1, kb, lane), acc1);
    acc2 = wmma_f16(af, load_bfrag(bp2, kb, lane), acc2);
    acc3 = wmma_f16(af, load_bfrag(bp3, kb, lane), acc3);
  }

  v8f accs[4] = {acc0, acc1, acc2, acc3};
  float lmax = 0.f;
#pragma unroll
  for (int nt = 0; nt < 4; ++nt) {
    const int n = cbase + nt * 16 + ml;
    const float inv = rsqrtf(var[n] + 1e-5f);
    const float ga = gamma[n], be = beta[n], mu = mean[n];
#pragma unroll
    for (int r = 0; r < 8; ++r) {
      int m = tm * 16 + grp * 8 + r;
      float v = (accs[nt][r] - mu) * inv * ga + be;
      v = fmaxf(v, 0.f);
      lmax = fmaxf(lmax, v);
      if (m < M) out[(size_t)m * N + n] = v;
    }
  }
  for (int o = 16; o; o >>= 1) lmax = fmaxf(lmax, __shfl_xor(lmax, o, 32));
  if (lane == 0) atomicMax(maxbits, __float_as_uint(lmax));
}

// ---------------------------------------------------------------------------
// Build streams: xs = concat(cls, conv/max) + pos  (writes f32 + f16)
// ---------------------------------------------------------------------------
__global__ void build_stream_kernel(const float* __restrict__ conv,  // [B,784,128]
                                    const float* __restrict__ pos,   // [S,128]
                                    const float* __restrict__ cls,   // [128]
                                    const float* __restrict__ maxv,
                                    float* __restrict__ s32,         // may be null
                                    half_t* __restrict__ s16) {
  int row = blockIdx.x;              // b*S + s
  int d = threadIdx.x;
  int b = row / SS, s = row % SS;
  float base = (s == 0) ? cls[d]
                        : conv[((size_t)b * PIX + (s - 1)) * DD + d] / maxv[0];
  float v = base + pos[(size_t)s * DD + d];
  if (s32) s32[(size_t)row * DD + d] = v;
  s16[(size_t)row * DD + d] = (half_t)v;
}

// ---------------------------------------------------------------------------
// Attention: one wave per (b, h, 16-query tile). Two-pass softmax in LDS.
// q,k: [B,H,S,DH] f16 ; vT: [B,H,DH,SP] f16 (zero padded s>=S) ; ctx: [B,S,512] f16
// dynamic LDS: 16*SP*4 (f32 scores) + 16*SP*2 (f16 probs) = 76800 bytes
// ---------------------------------------------------------------------------
__global__ void attn_kernel(const half_t* __restrict__ q,
                            const half_t* __restrict__ k,
                            const half_t* __restrict__ vT,
                            const int* __restrict__ enc,
                            half_t* __restrict__ ctx) {
  extern __shared__ char smem[];
  float*  sc = (float*)smem;                    // [16][SP]
  half_t* pb = (half_t*)(smem + 16 * SP * 4);   // [16][SP]

  const int lane = threadIdx.x;
  const int qt = blockIdx.x, h = blockIdx.y, b = blockIdx.z;
  const int ml = lane & 15, grp = lane >> 4;
  const float scale = 0.125f;                   // 1/sqrt(64)

  // Q fragments for this tile (K-dim = DH = 64 -> two 32-chunks)
  int qrow = qt * 16 + ml; if (qrow > SS - 1) qrow = SS - 1;
  const half_t* qp = q + ((size_t)(b * HH + h) * SS + qrow) * DH;
  v16h aq0 = load_afrag(qp, 0, lane);
  v16h aq1 = load_afrag(qp, 32, lane);

  // ---- scores = q @ k^T * scale, masked ----
  for (int kt = 0; kt < SP / 16; ++kt) {
    int key = kt * 16 + ml; int keyc = key > SS - 1 ? SS - 1 : key;
    const half_t* kp = k + ((size_t)(b * HH + h) * SS + keyc) * DH;
    v16h bf0 = load_bfrag(kp, 0, lane);
    v16h bf1 = load_bfrag(kp, 32, lane);
    v8f c = {};
    c = wmma_f16(aq0, bf0, c);
    c = wmma_f16(aq1, bf1, c);
    int kid = kt * 16 + ml;                     // lane's key column
    bool msk = (kid >= SS) || (enc[b * SS + (kid < SS ? kid : 0)] == 0);
#pragma unroll
    for (int r = 0; r < 8; ++r) {
      int m = grp * 8 + r;
      float v = c[r] * scale;
      if (msk) v = -1e30f;
      sc[m * SP + kid] = v;
    }
  }
  __syncthreads();

  // ---- softmax per query row (wave-cooperative) ----
  for (int m = 0; m < 16; ++m) {
    float mx = -1e30f;
    for (int c = lane; c < SP; c += 32) mx = fmaxf(mx, sc[m * SP + c]);
    for (int o = 16; o; o >>= 1) mx = fmaxf(mx, __shfl_xor(mx, o, 32));
    float sum = 0.f;
    for (int c = lane; c < SP; c += 32) {
      float e = __expf(sc[m * SP + c] - mx);
      sc[m * SP + c] = e;
      sum += e;
    }
    for (int o = 16; o; o >>= 1) sum += __shfl_xor(sum, o, 32);
    float inv = 1.f / sum;
    for (int c = lane; c < SP; c += 32)
      pb[m * SP + c] = (half_t)(sc[m * SP + c] * inv);
  }
  __syncthreads();

  // ---- ctx = probs @ v  (M=16 queries, N=DH=64, K=SP) ----
  v8f cacc[4] = {{}, {}, {}, {}};
  for (int kb = 0; kb < SP; kb += 32) {
    v16h af = load_afrag(pb + (size_t)ml * SP, kb, lane);
#pragma unroll
    for (int nt = 0; nt < 4; ++nt) {
      const half_t* vp = vT + ((size_t)(b * HH + h) * DH + nt * 16 + ml) * SP;
      cacc[nt] = wmma_f16(af, load_bfrag(vp, kb, lane), cacc[nt]);
    }
  }
#pragma unroll
  for (int nt = 0; nt < 4; ++nt)
#pragma unroll
    for (int r = 0; r < 8; ++r) {
      int s = qt * 16 + grp * 8 + r;
      if (s >= SS) continue;
      int dh = nt * 16 + ml;
      ctx[((size_t)b * SS + s) * (HH * DH) + h * DH + dh] = (half_t)cacc[nt][r];
    }
}

// ---------------------------------------------------------------------------
// Residual add + LayerNorm (eps=1e-12), one wave per token row (D=128)
// updates xs32 (f32 residual stream) and xs16 (f16 matmul copy)
// ---------------------------------------------------------------------------
__global__ void add_ln_kernel(const float* __restrict__ delta,
                              const float* __restrict__ g,
                              const float* __restrict__ bn,
                              float* __restrict__ xs32,
                              half_t* __restrict__ xs16) {
  int row = blockIdx.x * 4 + threadIdx.y;
  if (row >= MTOK) return;
  int lane = threadIdx.x;
  float v[4]; float sum = 0.f;
#pragma unroll
  for (int j = 0; j < 4; ++j) {
    int d = lane + 32 * j;
    v[j] = xs32[(size_t)row * DD + d] + delta[(size_t)row * DD + d];
    sum += v[j];
  }
  for (int o = 16; o; o >>= 1) sum += __shfl_xor(sum, o, 32);
  float mu = sum * (1.f / DD);
  float var = 0.f;
#pragma unroll
  for (int j = 0; j < 4; ++j) { float t = v[j] - mu; var += t * t; }
  for (int o = 16; o; o >>= 1) var += __shfl_xor(var, o, 32);
  float inv = rsqrtf(var * (1.f / DD) + 1e-12f);
#pragma unroll
  for (int j = 0; j < 4; ++j) {
    int d = lane + 32 * j;
    float o = (v[j] - mu) * inv * g[d] + bn[d];
    xs32[(size_t)row * DD + d] = o;
    xs16[(size_t)row * DD + d] = (half_t)o;
  }
}

// ---------------------------------------------------------------------------
// Utility kernels
// ---------------------------------------------------------------------------
__global__ void zero_u32_kernel(unsigned int* p, size_t n) {
  size_t i = (size_t)blockIdx.x * blockDim.x + threadIdx.x;
  if (i < n) p[i] = 0u;
}
__global__ void f32_to_f16_kernel(const float* src, half_t* dst, size_t n) {
  size_t i = (size_t)blockIdx.x * blockDim.x + threadIdx.x;
  if (i < n) dst[i] = (half_t)src[i];
}
// src f32 [R][C] -> dst f16 [C][R]
__global__ void transpose_f16_kernel(const float* src, half_t* dst, int R, int C) {
  size_t i = (size_t)blockIdx.x * blockDim.x + threadIdx.x;
  if (i >= (size_t)R * C) return;
  int c = (int)(i / R), r = (int)(i % R);
  dst[i] = (half_t)src[(size_t)r * C + c];
}
// x f32 [B,512,784] -> xT f16 [B,784,512]
__global__ void transpose_x_kernel(const float* x, half_t* xt) {
  size_t i = (size_t)blockIdx.x * blockDim.x + threadIdx.x;
  if (i >= (size_t)BB * PIX * CIN) return;
  int c = (int)(i % CIN);
  int p = (int)((i / CIN) % PIX);
  int b = (int)(i / ((size_t)CIN * PIX));
  xt[i] = (half_t)x[((size_t)b * CIN + c) * PIX + p];
}
// head1: hcls[b,n] = relu(sum_d xs32[b*S, d] * w1[d, n])
__global__ void head1_kernel(const float* __restrict__ xs32,
                             const float* __restrict__ w1,
                             float* __restrict__ hcls) {
  int b = blockIdx.x;
  int n = blockIdx.y * blockDim.x + threadIdx.x;
  float s = 0.f;
  const float* xr = xs32 + (size_t)b * SS * DD;   // cls row
  for (int d = 0; d < DD; ++d) s += xr[d] * w1[(size_t)d * 512 + n];
  hcls[(size_t)b * 512 + n] = fmaxf(s, 0.f);
}
// head2: out[b] = sum_n hcls[b,n] * w2[n]
__global__ void head2_kernel(const float* __restrict__ hcls,
                             const float* __restrict__ w2,
                             float* __restrict__ out) {
  __shared__ float red[256];
  int b = blockIdx.x, t = threadIdx.x;
  float s = 0.f;
  for (int n = t; n < 512; n += 256) s += hcls[(size_t)b * 512 + n] * w2[n];
  red[t] = s;
  __syncthreads();
  for (int o = 128; o; o >>= 1) {
    if (t < o) red[t] += red[t + o];
    __syncthreads();
  }
  if (t == 0) out[b] = red[0];
}

// ---------------------------------------------------------------------------
// Host orchestration
// ---------------------------------------------------------------------------
static inline size_t alignup(size_t x) { return (x + 255) & ~(size_t)255; }

extern "C" void kernel_launch(void* const* d_in, const int* in_sizes, int n_in,
                              void* d_out, int out_size, void* d_ws, size_t ws_size,
                              hipStream_t stream) {
  // ---- inputs (setup_inputs order) ----
  const float* conv_w  = (const float*)d_in[0];
  const float* bn_g    = (const float*)d_in[1];
  const float* bn_b    = (const float*)d_in[2];
  const float* bn_m    = (const float*)d_in[3];
  const float* bn_v    = (const float*)d_in[4];
  const float* pos     = (const float*)d_in[5];
  const float* cls     = (const float*)d_in[6];
  const float* Wq      = (const float*)d_in[7];
  const float* bq      = (const float*)d_in[8];
  const float* Wk      = (const float*)d_in[9];
  const float* bk      = (const float*)d_in[10];
  const float* Wv      = (const float*)d_in[11];
  const float* bv      = (const float*)d_in[12];
  const float* Wo      = (const float*)d_in[13];
  const float* bo      = (const float*)d_in[14];
  const float* ln1_g   = (const float*)d_in[15];
  const float* ln1_b   = (const float*)d_in[16];
  const float* fw1     = (const float*)d_in[17];
  const float* fb1     = (const float*)d_in[18];
  const float* fw2     = (const float*)d_in[19];
  const float* fb2     = (const float*)d_in[20];
  const float* ln2_g   = (const float*)d_in[21];
  const float* ln2_b   = (const float*)d_in[22];
  const float* pw1     = (const float*)d_in[23];
  const float* pw2     = (const float*)d_in[24];
  const int*   enc     = (const int*)d_in[25];
  const float* xin     = (const float*)d_in[26];
  const float* yin     = (const float*)d_in[27];
  float* out = (float*)d_out;

  // ---- workspace carving ----
  char* w = (char*)d_ws; size_t off = 0;
  auto carve = [&](size_t bytes) { char* p = w + off; off = alignup(off + bytes); return p; };
  unsigned int* maxv  = (unsigned int*)carve(4);
  half_t* xT   = (half_t*)carve((size_t)BB * PIX * CIN * 2);
  half_t* yT   = (half_t*)carve((size_t)BB * PIX * CIN * 2);
  float*  xc   = (float*) carve((size_t)BB * PIX * DD * 4);
  float*  yc   = (float*) carve((size_t)BB * PIX * DD * 4);
  float*  xs32 = (float*) carve((size_t)MTOK * DD * 4);
  half_t* xs16 = (half_t*)carve((size_t)MTOK * DD * 2);
  half_t* xf16 = (half_t*)carve((size_t)MTOK * DD * 2);
  half_t* cw16 = (half_t*)carve((size_t)DD * CIN * 2);
  half_t* WqT  = (half_t*)carve((size_t)LL * 512 * DD * 2);
  half_t* WkT  = (half_t*)carve((size_t)LL * 512 * DD * 2);
  half_t* WvT  = (half_t*)carve((size_t)LL * 512 * DD * 2);
  half_t* WoT  = (half_t*)carve((size_t)LL * DD * 512 * 2);
  half_t* W1T  = (half_t*)carve((size_t)LL * 512 * DD * 2);
  half_t* W2T  = (half_t*)carve((size_t)LL * DD * 512 * 2);
  half_t* q16  = (half_t*)carve((size_t)BB * HH * SS * DH * 2);
  half_t* k16  = (half_t*)carve((size_t)BB * HH * SS * DH * 2);
  half_t* vT16 = (half_t*)carve((size_t)BB * HH * DH * SP * 2);
  half_t* ctx16= (half_t*)carve((size_t)MTOK * 512 * 2);    // reused as FFN hidden
  float*  del32= (float*) carve((size_t)MTOK * DD * 4);
  float*  hcls = (float*) carve((size_t)BB * 512 * 4);
  (void)ws_size; (void)n_in; (void)in_sizes; (void)out_size;

  const dim3 wblk(32, 4);

  // ---- init: zero maxval and vT padding buffer ----
  zero_u32_kernel<<<1, 32, 0, stream>>>(maxv, 1);
  {
    size_t n = (size_t)BB * HH * DH * SP / 2;   // halves -> u32 count
    zero_u32_kernel<<<(unsigned)((n + 255) / 256), 256, 0, stream>>>((unsigned int*)vT16, n);
  }

  // ---- convert inputs / weights to f16 (transposed for B-fragments) ----
  {
    size_t n = (size_t)BB * PIX * CIN;
    transpose_x_kernel<<<(unsigned)((n + 255) / 256), 256, 0, stream>>>(xin, xT);
    transpose_x_kernel<<<(unsigned)((n + 255) / 256), 256, 0, stream>>>(yin, yT);
  }
  f32_to_f16_kernel<<<(DD * CIN + 255) / 256, 256, 0, stream>>>(conv_w, cw16, (size_t)DD * CIN); // already [N=128][K=512]
  for (int i = 0; i < LL; ++i) {
    const size_t msz = (size_t)DD * 512;            // 65536 elements each
    unsigned g = (unsigned)((msz + 255) / 256);
    transpose_f16_kernel<<<g, 256, 0, stream>>>(Wq + i * msz, WqT + i * msz, DD, 512);
    transpose_f16_kernel<<<g, 256, 0, stream>>>(Wk + i * msz, WkT + i * msz, DD, 512);
    transpose_f16_kernel<<<g, 256, 0, stream>>>(Wv + i * msz, WvT + i * msz, DD, 512);
    transpose_f16_kernel<<<g, 256, 0, stream>>>(Wo + i * msz, WoT + i * msz, 512, DD);
    transpose_f16_kernel<<<g, 256, 0, stream>>>(fw1 + i * msz, W1T + i * msz, DD, 512);
    transpose_f16_kernel<<<g, 256, 0, stream>>>(fw2 + i * msz, W2T + i * msz, 512, DD);
  }

  // ---- conv stem (both streams) + global max ----
  {
    dim3 grid(BB * PIX / 16, 1);        // wave strips cover N=128 (2 active waves)
    stem_gemm_kernel<<<grid, wblk, 0, stream>>>(xT, cw16, bn_g, bn_b, bn_m, bn_v, xc, maxv);
    stem_gemm_kernel<<<grid, wblk, 0, stream>>>(yT, cw16, bn_g, bn_b, bn_m, bn_v, yc, maxv);
  }

  // ---- build streams: xs (K,V evolving) from x, xf (Q fixed) from y ----
  build_stream_kernel<<<MTOK, DD, 0, stream>>>(xc, pos, cls, (const float*)maxv, xs32, xs16);
  build_stream_kernel<<<MTOK, DD, 0, stream>>>(yc, pos, cls, (const float*)maxv, nullptr, xf16);

  // ---- transformer layers ----
  const int MT = (MTOK + 15) / 16;   // 393
  for (int i = 0; i < LL; ++i) {
    const half_t* wqt = WqT + (size_t)i * 512 * DD;
    const half_t* wkt = WkT + (size_t)i * 512 * DD;
    const half_t* wvt = WvT + (size_t)i * 512 * DD;
    const half_t* wot = WoT + (size_t)i * DD * 512;
    const half_t* w1t = W1T + (size_t)i * 512 * DD;
    const half_t* w2t = W2T + (size_t)i * DD * 512;

    // Q/K/V projections (N=512, K=128): grid.y = 512/256 = 2
    dim3 g512(MT, 2);
    gemm_wmma_kernel<<<g512, wblk, 0, stream>>>(xf16, wqt, bq + i * 512, nullptr, q16,
                                                MTOK, 512, DD, 0, 2);
    gemm_wmma_kernel<<<g512, wblk, 0, stream>>>(xs16, wkt, bk + i * 512, nullptr, k16,
                                                MTOK, 512, DD, 0, 2);
    gemm_wmma_kernel<<<g512, wblk, 0, stream>>>(xs16, wvt, bv + i * 512, nullptr, vT16,
                                                MTOK, 512, DD, 0, 3);

    // attention
    {
      dim3 ag(SP / 16, HH, BB);
      size_t shmem = (size_t)16 * SP * 4 + (size_t)16 * SP * 2;
      attn_kernel<<<ag, 32, shmem, stream>>>(q16, k16, vT16, enc, ctx16);
    }

    // output projection (N=128, K=512) -> delta ; grid.y = 1 (2 active waves)
    dim3 g128(MT, 1);
    gemm_wmma_kernel<<<g128, wblk, 0, stream>>>(ctx16, wot, bo + i * DD, del32, nullptr,
                                                MTOK, DD, 512, 0, 0);
    add_ln_kernel<<<(MTOK + 3) / 4, wblk, 0, stream>>>(del32, ln1_g + i * DD, ln1_b + i * DD,
                                                       xs32, xs16);

    // FFN: GELU(xs @ w1 + b1) @ w2 + b2 ; hidden reuses ctx16
    gemm_wmma_kernel<<<g512, wblk, 0, stream>>>(xs16, w1t, fb1 + i * 512, nullptr, ctx16,
                                                MTOK, 512, DD, 1, 1);
    gemm_wmma_kernel<<<g128, wblk, 0, stream>>>(ctx16, w2t, fb2 + i * DD, del32, nullptr,
                                                MTOK, DD, 512, 0, 0);
    add_ln_kernel<<<(MTOK + 3) / 4, wblk, 0, stream>>>(del32, ln2_g + i * DD, ln2_b + i * DD,
                                                       xs32, xs16);
  }

  // ---- MLP head (f32, tiny) ----
  head1_kernel<<<dim3(BB, 4), 128, 0, stream>>>(xs32, pw1, hcls);
  head2_kernel<<<BB, 256, 0, stream>>>(hcls, pw2, out);
}
